// DigitMoE_20100446946124
// MI455X (gfx1250) — matrix operand order repaired and checked by reading, not compiled, not run on hardware
//
#include <hip/hip_runtime.h>

typedef __attribute__((ext_vector_type(16))) _Float16 v16h;
typedef __attribute__((ext_vector_type(8)))  _Float16 v8h;
typedef __attribute__((ext_vector_type(8)))  float    v8f;
typedef __attribute__((ext_vector_type(4)))  unsigned v4u;
typedef __attribute__((ext_vector_type(4)))  float    v4f;

#define B_TOTAL 65536
#define D_INF   784
#define D_PAD   800     // 25 * 32
#define H1V     256
#define H2V     64
#define NE      10
#define BM      64      // batch rows per workgroup
#define WBANK_H 8192    // halves per weight bank (32 k x 256 n)

// LDS layout (byte offsets), all 16B aligned
#define XOFF 0
#define WOFF (BM * D_PAD * 2)            // 102400: x tile f16 [64][800]
#define HOFF (WOFF + 32768)              // 135168: w stage, 2 banks (or 32KB for W2)
#define EOFF (HOFF + BM * H1V * 2)       // 167936: h tile f16 [64][256]
#define GOFF (EOFF + BM * NE * 4)        // 170496: expert outputs f32 [64][10]
#define SMEM_BYTES (GOFF + BM * NE * 4)  // 173056: gate logits f32 [64][10]

// A fragment (16x32 f16): lanes 0-15 -> rows m0..m0+15, K k0..k0+15;
// lanes 16-31 same rows, K k0+16..k0+31.
__device__ __forceinline__ v16h frag_a(const _Float16* base, int stride, int m0,
                                       int k0, int lane) {
  const int row = m0 + (lane & 15);
  const int h  = (lane >> 4) & 1;
  const _Float16* p = base + row * stride + k0 + h * 16;
  v8h lo = *(const v8h*)p;
  v8h hi = *(const v8h*)(p + 8);
  return __builtin_shufflevector(lo, hi, 0,1,2,3,4,5,6,7,8,9,10,11,12,13,14,15);
}

// B fragment (32x16 f16) from column-major LDS stage [n][K]:
// lanes 0-15: col n0+lane, K k0..k0+15; lanes 16-31: col n0+lane-16, K k0+16..k0+31.
__device__ __forceinline__ v16h frag_b(const _Float16* base, int stride, int n0,
                                       int k0, int lane) {
  const _Float16* p = base + (n0 + (lane & 15)) * stride + k0 + ((lane >> 4) & 1) * 16;
  v8h lo = *(const v8h*)p;
  v8h hi = *(const v8h*)(p + 8);
  return __builtin_shufflevector(lo, hi, 0,1,2,3,4,5,6,7,8,9,10,11,12,13,14,15);
}

__device__ __forceinline__ unsigned pack2(float a, float b) {
  union { _Float16 h[2]; unsigned u; } p;
  p.h[0] = (_Float16)a; p.h[1] = (_Float16)b;
  return p.u;
}

// Fetch one 32xH1 weight window (column n = tid, rows k0..k0+31) as raw f32
// into registers. No dependent ALU here: loads can issue as one clause and the
// loadcnt wait lands at consumption time (in store_window, after the WMMAs).
__device__ __forceinline__ void fetch_window(const float* __restrict__ Wsrc,
                                             int k0, int tid, float* wf) {
  const float* col = Wsrc + tid;
  if (k0 + 32 <= D_INF) {
#pragma unroll
    for (int i = 0; i < 32; ++i) wf[i] = col[(size_t)(k0 + i) * H1V];
  } else {
#pragma unroll
    for (int i = 0; i < 32; ++i) {
      const int k = k0 + i;
      float v = col[(size_t)(k < D_INF ? k : 0) * H1V];  // clamped, always safe
      wf[i] = (k < D_INF) ? v : 0.0f;
    }
  }
}

// Convert + pack + store the fetched window into an LDS bank (f16, col-major).
__device__ __forceinline__ void store_window(_Float16* bank, int tid,
                                             const float* wf) {
  v4u* wp = (v4u*)bank + tid * 4;   // column n=tid -> halves [n*32, n*32+32)
#pragma unroll
  for (int q = 0; q < 4; ++q) {
    v4u t = {pack2(wf[8 * q + 0], wf[8 * q + 1]),
             pack2(wf[8 * q + 2], wf[8 * q + 3]),
             pack2(wf[8 * q + 4], wf[8 * q + 5]),
             pack2(wf[8 * q + 6], wf[8 * q + 7])};
    wp[q] = t;
  }
}

__global__ __launch_bounds__(256)
void moe_fused_kernel(const float* __restrict__ xg,
                      const float* __restrict__ W1g, const float* __restrict__ b1g,
                      const float* __restrict__ W2g, const float* __restrict__ b2g,
                      const float* __restrict__ W3g, const float* __restrict__ b3g,
                      const float* __restrict__ Wg1g, const float* __restrict__ bg1g,
                      const float* __restrict__ Wg2g, const float* __restrict__ bg2g,
                      float* __restrict__ outg) {
  extern __shared__ __attribute__((aligned(16))) char smem[];
  _Float16* xlds = (_Float16*)(smem + XOFF);
  _Float16* wlds = (_Float16*)(smem + WOFF);
  _Float16* hlds = (_Float16*)(smem + HOFF);
  float*    eout = (float*)(smem + EOFF);
  float*    glds = (float*)(smem + GOFF);

  const int tid  = threadIdx.x;
  const int wave = tid >> 5;
  const int lane = tid & 31;
  const int bg   = blockIdx.x * BM;

  // ---- stage x tile: float4 loads -> packed f16 b64 stores, K padded to 800 ----
  for (int idx = tid; idx < BM * (D_PAD / 4); idx += 256) {
    const int row = idx / (D_PAD / 4);
    const int c4  = idx - row * (D_PAD / 4);       // 0..199, valid < 196
    const v4f* src = (const v4f*)(xg + (size_t)(bg + row) * D_INF);
    v4f v = src[c4 < (D_INF / 4) ? c4 : 0];
    const float msk = (c4 < (D_INF / 4)) ? 1.0f : 0.0f;
    unsigned* dst = (unsigned*)(xlds + row * D_PAD + c4 * 4);
    dst[0] = pack2(v.x * msk, v.y * msk);
    dst[1] = pack2(v.z * msk, v.w * msk);
  }

  const int m0a   = (wave & 3) * 16;   // wave's M band for layer 1
  const int nbase = (wave >> 2) * 128; // wave's N half for layer 1

  for (int j = 0; j <= NE; ++j) {      // experts 0..9, then gate layer as j==10
    const float* Wsrc = (j < NE) ? (W1g + (size_t)j * D_INF * H1V) : Wg1g;
    const float* bsrc = (j < NE) ? (b1g + j * H1V) : bg1g;

    // warm L2 for this expert's layer-2 weights while layer-1 runs
    if (j < NE)
      __builtin_prefetch(W2g + (size_t)j * H1V * H2V + (size_t)tid * 64, 0, 1);

    v8f acc[8];
#pragma unroll
    for (int t = 0; t < 8; ++t) acc[t] = (v8f){0, 0, 0, 0, 0, 0, 0, 0};

    // ---- layer 1: [64,800] x [800,256], K-step 32, double-buffered weights ----
    float wf[32];
    fetch_window(Wsrc, 0, tid, wf);
    store_window(wlds, tid, wf);       // bank 0 (prior readers fenced last iter)
    __syncthreads();

    for (int ks = 0; ks < 25; ++ks) {
      const int k0 = ks * 32;
      const _Float16* wbank = wlds + (ks & 1) * WBANK_H;
      if (ks < 24) fetch_window(Wsrc, k0 + 32, tid, wf);  // overlap with WMMA

      v16h a = frag_a(xlds, D_PAD, m0a, k0, lane);
#pragma unroll
      for (int g = 0; g < 2; ++g) {
        v16h bb[4];
#pragma unroll
        for (int t = 0; t < 4; ++t)
          bb[t] = frag_b(wbank, 32, nbase + (g * 4 + t) * 16, 0, lane);
#pragma unroll
        for (int t = 0; t < 4; ++t)
          acc[g * 4 + t] = __builtin_amdgcn_wmma_f32_16x16x32_f16(
              false, a, false, bb[t], (short)0, acc[g * 4 + t], false, false);
      }

      if (ks < 24) store_window(wlds + ((ks + 1) & 1) * WBANK_H, tid, wf);
      __syncthreads();
    }

    // ---- bias + relu, h -> LDS f16 [64][256] ----
    {
      const int rbase = m0a + ((lane >> 4) & 1) * 8;
#pragma unroll
      for (int t = 0; t < 8; ++t) {
        const int n = nbase + t * 16 + (lane & 15);
        const float bn = bsrc[n];
#pragma unroll
        for (int r = 0; r < 8; ++r) {
          float v = acc[t][r] + bn;
          v = v > 0.0f ? v : 0.0f;
          hlds[(rbase + r) * H1V + n] = (_Float16)v;
        }
      }
    }
    __syncthreads();  // h visible; wlds free for reuse

    if (j < NE) {
      // ---- stage W2[j] (256x64 f32) as f16 column-major [n=64][k=256] ----
      const float* W2src = W2g + (size_t)j * H1V * H2V;
#pragma unroll 4
      for (int i = 0; i < 64; ++i) {
        const int idx = i * 256 + tid;
        const int k = idx >> 6;
        const int n = idx & 63;
        wlds[n * H1V + k] = (_Float16)W2src[k * H2V + n];
      }
      __syncthreads();

      // ---- layer 2 + layer 3 fused (waves 0..3) ----
      if (wave < 4) {
        const int m0 = wave * 16;
        v8f a2[4];
#pragma unroll
        for (int t = 0; t < 4; ++t) a2[t] = (v8f){0, 0, 0, 0, 0, 0, 0, 0};

        for (int k0 = 0; k0 < H1V; k0 += 32) {
          v16h a = frag_a(hlds, H1V, m0, k0, lane);
          v16h bb[4];
#pragma unroll
          for (int t = 0; t < 4; ++t)
            bb[t] = frag_b(wlds, H1V, t * 16, k0, lane);
#pragma unroll
          for (int t = 0; t < 4; ++t)
            a2[t] = __builtin_amdgcn_wmma_f32_16x16x32_f16(
                false, a, false, bb[t], (short)0, a2[t], false, false);
        }

        // relu(h2 + b2) . W3, reduced over N=64
        float partial[8];
#pragma unroll
        for (int r = 0; r < 8; ++r) partial[r] = 0.0f;
#pragma unroll
        for (int t = 0; t < 4; ++t) {
          const int n = t * 16 + (lane & 15);
          const float bn = b2g[j * H2V + n];
          const float w3 = W3g[j * H2V + n];
#pragma unroll
          for (int r = 0; r < 8; ++r) {
            float v = a2[t][r] + bn;
            v = v > 0.0f ? v : 0.0f;
            partial[r] += v * w3;
          }
        }
#pragma unroll
        for (int r = 0; r < 8; ++r) {
#pragma unroll
          for (int m = 1; m < 16; m <<= 1)
            partial[r] += __shfl_xor(partial[r], m, 16);
        }
        if ((lane & 15) == 0) {
          const int rbase = m0 + ((lane >> 4) & 1) * 8;
          const float b3v = b3g[j];
#pragma unroll
          for (int r = 0; r < 8; ++r)
            eout[(rbase + r) * NE + j] = partial[r] + b3v;
        }
      }
      __syncthreads();
    }
  }

  // ---- gate head: stage Wg2 (256x10 f32) into LDS, then logits ----
  {
    float* wg2l = (float*)wlds;  // 2560 floats = 10KB
    for (int idx = tid; idx < H1V * NE; idx += 256) wg2l[idx] = Wg2g[idx];
    __syncthreads();
    for (int idx = tid; idx < BM * NE; idx += 256) {
      const int row = idx / NE;
      const int e   = idx - row * NE;
      float s = bg2g[e];
      for (int k = 0; k < H1V; ++k)
        s += (float)hlds[row * H1V + k] * wg2l[k * NE + e];
      glds[idx] = s;
    }
  }
  __syncthreads();

  // ---- softmax + combine, write all three outputs ----
  if (tid < BM) {
    const int row = tid;
    float mx = -1e30f;
    for (int e = 0; e < NE; ++e) mx = fmaxf(mx, glds[row * NE + e]);
    float ex[NE];
    float sum = 0.0f;
    for (int e = 0; e < NE; ++e) {
      ex[e] = __expf(glds[row * NE + e] - mx);
      sum += ex[e];
    }
    const float inv = 1.0f / sum;
    const size_t bo = (size_t)(bg + row) * NE;
    const size_t sz = (size_t)B_TOTAL * NE;
    for (int e = 0; e < NE; ++e) {
      const float gs = ex[e] * inv;
      const float eo = eout[row * NE + e];
      outg[bo + e]          = gs * eo;  // output
      outg[sz + bo + e]     = gs;       // gate_scores
      outg[2 * sz + bo + e] = eo;       // expert_outputs
    }
  }
}

extern "C" void kernel_launch(void* const* d_in, const int* in_sizes, int n_in,
                              void* d_out, int out_size, void* d_ws, size_t ws_size,
                              hipStream_t stream) {
  (void)in_sizes; (void)n_in; (void)d_ws; (void)ws_size; (void)out_size;
  const float* xg   = (const float*)d_in[0];
  const float* W1g  = (const float*)d_in[1];
  const float* b1g  = (const float*)d_in[2];
  const float* W2g  = (const float*)d_in[3];
  const float* b2g  = (const float*)d_in[4];
  const float* W3g  = (const float*)d_in[5];
  const float* b3g  = (const float*)d_in[6];
  const float* Wg1g = (const float*)d_in[7];
  const float* bg1g = (const float*)d_in[8];
  const float* Wg2g = (const float*)d_in[9];
  const float* bg2g = (const float*)d_in[10];
  float* outg = (float*)d_out;

  (void)hipFuncSetAttribute((const void*)moe_fused_kernel,
                            hipFuncAttributeMaxDynamicSharedMemorySize,
                            (int)SMEM_BYTES);
  dim3 grid(B_TOTAL / BM);
  dim3 block(256);
  moe_fused_kernel<<<grid, block, SMEM_BYTES, stream>>>(
      xg, W1g, b1g, W2g, b2g, W3g, b3g, Wg1g, bg1g, Wg2g, bg2g, outg);
}
